// AttentionGuidedInterpolation_86663850099178
// MI455X (gfx1250) — matrix operand order repaired and checked by reading, not compiled, not run on hardware
//
#include <hip/hip_runtime.h>
#include <hip/hip_bf16.h>

#define NB 4
#define CC 128
#define DD 16
#define HH 32
#define WW 32
#define KK 8192
#define LL 1024
#define TOPK 5
#define ANB 9
#define NHD 8
#define HD 16

typedef __attribute__((ext_vector_type(16))) _Float16 v16h;
typedef __attribute__((ext_vector_type(8)))  float    v8f;
typedef __attribute__((ext_vector_type(2)))  float    v2f;

// s_wait_asynccnt: builtin if the toolchain declares it, else raw ISA
#if defined(__has_builtin)
# if __has_builtin(__builtin_amdgcn_s_wait_asynccnt)
#  define WAIT_ASYNC(n) __builtin_amdgcn_s_wait_asynccnt(n)
# endif
#endif
#ifndef WAIT_ASYNC
# define WAIT_ASYNC(n) asm volatile("s_wait_asynccnt %0" :: "i"(n) : "memory")
#endif

// sl(s,n)[c][j] view of feature_map (N,C,D,H,W): slice s along D, j indexes H*W
__device__ __forceinline__ float slv(const float* fm, int n, int s, int c, int j) {
  return fm[(((n * CC + c) * DD + s) << 10) + j];
}

// ---------------------------------------------------------------------------
// Kernel 1: fused Gram (f32 WMMA) + top-5 + distance-weighted gather.
// B operand is staged global->LDS with GLOBAL_LOAD_ASYNC_TO_LDS_B128,
// double-buffered in 64-column chunks and overlapped with WMMA compute.
// grid: (32 row-blocks, 64 (n,s) pairs), block 256 (8 waves)
// ---------------------------------------------------------------------------
#define BROW 80   // padded LDS row stride (floats): +16 keeps lane halves on
                  // disjoint banks (2*80 mod 64 = 32)

__global__ __launch_bounds__(256) void sim_search_kernel(const float* __restrict__ fm,
                                                         float* __restrict__ simf) {
  extern __shared__ float lds[];
  float* sim  = lds;                               // 32*1024 f (128 KB)
  float* Bbuf = lds + 32 * 1024;                   // 2 * 128*BROW f (80 KB)
  float* cv   = Bbuf + 2 * 128 * BROW;             // 32*8*5 candidate values
  int*   ci   = (int*)(cv + 32 * 8 * 5);           // 32*8*5 candidate indices
  int*   fi   = (int*)(ci + 32 * 8 * 5);           // 32*5 final indices
  float* fw   = (float*)(fi + 32 * 5);             // 32*5 final weights

  const int p    = blockIdx.y;                     // 0..63
  const int n    = p >> 4;
  const int s    = p & 15;
  const int r0g  = blockIdx.x * 32;                // global row block base
  const int tid  = threadIdx.x;
  const int wave = tid >> 5;
  const int lane = tid & 31;
  const int lm   = lane & 15;
  const int hi   = lane >> 4;                      // 0: lanes 0-15, 1: lanes 16-31

  // --- async copy of one 64-column B chunk (128 x 64 f32 = 32 KB) ---
  // 2048 16B segments; 8 per thread; 8 async instructions per wave per chunk
  auto issue_chunk = [&](int ch, int buf) {
    const int jbase = ch * 64;
    float* bb = Bbuf + buf * (128 * BROW);
#pragma unroll
    for (int r = 0; r < 8; ++r) {
      const int i  = tid + 256 * r;
      const int c  = i >> 4;                       // 0..127 channel row
      const int sg = i & 15;                       // 16B segment in row
      const float* gp = fm + (((size_t)((n * CC + c) * DD + s)) << 10) + jbase + sg * 4;
      const unsigned ldsoff = (unsigned)(uintptr_t)(bb + c * BROW + sg * 4);
      asm volatile("global_load_async_to_lds_b128 %0, %1, off"
                   :: "v"(ldsoff), "v"(gp) : "memory");
    }
  };

  // --- Gram block: 32 rows x 1024 cols via v_wmma_f32_16x16x4_f32 ---
  const int rs    = wave & 1;                      // row strip (16 rows)
  const int wq    = wave >> 1;                     // column tile within chunk
  const int rbase = r0g + rs * 16;

  issue_chunk(0, 0);

  // Preload A fragments for all 32 K-chunks (K=4 each over C=128)
  // layout: v0 = A[m][K0] (lanes 0-15) / A[m][K2] (lanes 16-31); v1 = K1 / K3
  v2f af[32];
#pragma unroll
  for (int kk = 0; kk < 32; ++kk) {
    const int c0 = kk * 4 + hi * 2;
    af[kk].x = slv(fm, n, s, c0,     rbase + lm);
    af[kk].y = slv(fm, n, s, c0 + 1, rbase + lm);
  }

  for (int ch = 0; ch < 16; ++ch) {
    if (ch < 15) {
      issue_chunk(ch + 1, (ch + 1) & 1);
      WAIT_ASYNC(8);          // my 8 copies of chunk `ch` have landed
    } else {
      WAIT_ASYNC(0);
    }
    __syncthreads();          // all waves' copies of chunk `ch` visible

    const float* bb  = Bbuf + (ch & 1) * (128 * BROW);
    const int jbase  = ch * 64;
    const int jc     = wq * 16 + lm;               // column within chunk
    v8f acc = {};
#pragma unroll
    for (int kk = 0; kk < 32; ++kk) {
      const int c0 = kk * 4 + hi * 2;
      v2f bf;
      bf.x = bb[c0 * BROW + jc];
      bf.y = bb[(c0 + 1) * BROW + jc];
      acc = __builtin_amdgcn_wmma_f32_16x16x4_f32(false, af[kk], false, bf,
                                                  (short)0, acc, false, false);
    }
#pragma unroll
    for (int vv = 0; vv < 8; ++vv) {
      const int m = vv + hi * 8;                   // C/D layout: VGPR vv -> rows vv / vv+8
      sim[(rs * 16 + m) * 1024 + jbase + wq * 16 + lm] = acc[vv];
    }
    __syncthreads();          // chunk consumed before its buffer is reused
  }

  // --- partial top-5: 8 threads per row, 128 cols each ---
  {
    const int row = tid >> 3;
    const int sub = tid & 7;
    float tv[5] = {-1e30f, -1e30f, -1e30f, -1e30f, -1e30f};
    int   ti[5] = {0, 0, 0, 0, 0};
    const float* rowp = sim + row * 1024 + sub * 128;
    for (int j = 0; j < 128; ++j) {
      const float x = rowp[j];
      if (x > tv[4]) {
        int pp = 4;
        while (pp > 0 && x > tv[pp - 1]) { tv[pp] = tv[pp - 1]; ti[pp] = ti[pp - 1]; --pp; }
        tv[pp] = x; ti[pp] = sub * 128 + j;
      }
    }
    const int b = (row * 8 + sub) * 5;
#pragma unroll
    for (int t = 0; t < 5; ++t) { cv[b + t] = tv[t]; ci[b + t] = ti[t]; }
  }
  __syncthreads();

  // --- merge + weights (1 thread per row) ---
  if (tid < 32) {
    const int row = tid;
    float tv[5] = {-1e30f, -1e30f, -1e30f, -1e30f, -1e30f};
    int   ti[5] = {0, 0, 0, 0, 0};
    for (int c8 = 0; c8 < 40; ++c8) {
      const float x  = cv[row * 40 + c8];
      const int   jj = ci[row * 40 + c8];
      if (x > tv[4]) {
        int pp = 4;
        while (pp > 0 && x > tv[pp - 1]) { tv[pp] = tv[pp - 1]; ti[pp] = ti[pp - 1]; --pp; }
        tv[pp] = x; ti[pp] = jj;
      }
    }
    const int ri = r0g + row;
    float wv[5], wsum = 0.f;
#pragma unroll
    for (int t = 0; t < 5; ++t) {
      wv[t] = 1.0f / (fabsf((float)(ti[t] - ri)) + 1e-5f);
      wsum += wv[t];
    }
#pragma unroll
    for (int t = 0; t < 5; ++t) { fw[row * 5 + t] = wv[t] / wsum; fi[row * 5 + t] = ti[t]; }
  }
  __syncthreads();

  // --- weighted gather -> sim_feats ---
  {
    const int row = tid >> 3;
    const int cg  = (tid & 7) * 16;
    const int ri  = r0g + row;
    int   idl[5]; float wl[5];
#pragma unroll
    for (int t = 0; t < 5; ++t) { idl[t] = fi[row * 5 + t]; wl[t] = fw[row * 5 + t]; }
    for (int c = cg; c < cg + 16; ++c) {
      float acc = 0.f;
#pragma unroll
      for (int t = 0; t < 5; ++t) acc += wl[t] * slv(fm, n, s, c, idl[t]);
      simf[(((n * CC + c) * DD + s) << 10) + ri] = acc;
    }
  }
}

// ---------------------------------------------------------------------------
// Kernel 2: trilinear init_fv + 9-neighbor nearest comb
// grid (K, N), block 128 (thread = channel)
// ---------------------------------------------------------------------------
__global__ __launch_bounds__(128) void sample_kernel(const float* __restrict__ fm,
                                                     const float* __restrict__ simf,
                                                     const float* __restrict__ xyz,
                                                     float* __restrict__ initfv,
                                                     float* __restrict__ comb) {
  const int k = blockIdx.x;
  const int n = blockIdx.y;
  const int c = threadIdx.x;
  const float x0c = xyz[(n * KK + k) * 3 + 0];   // -> D
  const float x1c = xyz[(n * KK + k) * 3 + 1];   // -> H
  const float x2c = xyz[(n * KK + k) * 3 + 2];   // -> W

  // bilinear sample (grid = xyz[..., ::-1])
  const float ix = ((x2c + 1.f) * WW - 1.f) * 0.5f;
  const float iy = ((x1c + 1.f) * HH - 1.f) * 0.5f;
  const float iz = ((x0c + 1.f) * DD - 1.f) * 0.5f;
  const float fx = floorf(ix), fy = floorf(iy), fz = floorf(iz);
  const float tx = ix - fx, ty = iy - fy, tz = iz - fz;
  const int x0 = (int)fx, y0 = (int)fy, z0 = (int)fz;
  const float* fmc = fm + (size_t)(n * CC + c) * DD * LL;
  float init = 0.f;
#pragma unroll
  for (int dz = 0; dz < 2; ++dz)
#pragma unroll
    for (int dy = 0; dy < 2; ++dy)
#pragma unroll
      for (int dx = 0; dx < 2; ++dx) {
        const int zz = z0 + dz, yy = y0 + dy, xx = x0 + dx;
        const bool ok = (zz >= 0) && (zz < DD) && (yy >= 0) && (yy < HH) &&
                        (xx >= 0) && (xx < WW);
        const float w = (dz ? tz : 1.f - tz) * (dy ? ty : 1.f - ty) * (dx ? tx : 1.f - tx);
        if (ok) init += fmc[zz * LL + yy * WW + xx] * w;
      }

  // neighbor coords (mdi = 0: fixed D, +-1 steps scaled by 2/H, 2/W)
  const float gd = floorf((x0c + 1.f) * 0.5f * 15.f);
  const float gh = floorf((x1c + 1.f) * 0.5f * 31.f);
  const float gw = floorf((x2c + 1.f) * 0.5f * 31.f);
  float rw[ANB], rwsum = 0.f;
  int vzi[ANB], vyi[ANB], vxi[ANB]; bool okn[ANB];
#pragma unroll
  for (int a = 0; a < ANB; ++a) {
    const float sh = (float)(a / 3 - 1) * (2.0f / HH);
    const float sv = (float)(a % 3 - 1) * (2.0f / WW);
    const float nd = gd / 15.f * 2.f - 1.f;
    const float nh = (gh + sh) / 31.f * 2.f - 1.f;
    const float nw = (gw + sv) / 31.f * 2.f - 1.f;
    const float d0 = x0c - nd, d1 = x1c - nh, d2 = x2c - nw;
    rw[a] = 1.f / (sqrtf(d0 * d0 + d1 * d1 + d2 * d2) + 1e-6f);
    rwsum += rw[a];
    const int vx = (int)rintf(((nw + 1.f) * WW - 1.f) * 0.5f);
    const int vy = (int)rintf(((nh + 1.f) * HH - 1.f) * 0.5f);
    const int vz = (int)rintf(((nd + 1.f) * DD - 1.f) * 0.5f);
    okn[a] = (vz >= 0) && (vz < DD) && (vy >= 0) && (vy < HH) && (vx >= 0) && (vx < WW);
    vzi[a] = vz; vyi[a] = vy; vxi[a] = vx;
  }
  const float* sfc = simf + (size_t)(n * CC + c) * DD * LL;
  const float inv = 1.f / rwsum;
  float acc = 0.f;
#pragma unroll
  for (int a = 0; a < ANB; ++a) {
    if (okn[a]) {
      const int off = vzi[a] * LL + vyi[a] * WW + vxi[a];
      acc += (fmc[off] + sfc[off]) * (rw[a] * inv);
    }
  }
  const int o = (n * KK + k) * CC + c;
  initfv[o] = init;
  comb[o]   = acc * 0.5f;
}

// ---------------------------------------------------------------------------
// Kernel 3: precompose affine chains  Wc = P_z @ W_z,  bc = P_z @ b_z + bp_z
// ---------------------------------------------------------------------------
__global__ __launch_bounds__(128) void compose_kernel(
    const float* __restrict__ Wq, const float* __restrict__ bq,
    const float* __restrict__ Wk, const float* __restrict__ bk,
    const float* __restrict__ Wv, const float* __restrict__ bv,
    const float* __restrict__ ipw, const float* __restrict__ ipb,
    float* __restrict__ Wc, float* __restrict__ bc) {
  const int which = blockIdx.x >> 7;   // 0..2
  const int i     = blockIdx.x & 127;
  const int j     = threadIdx.x;
  const float* Wsel = (which == 0) ? Wq : (which == 1) ? Wk : Wv;
  const float* bsel = (which == 0) ? bq : (which == 1) ? bk : bv;
  const float* P    = ipw + which * CC * CC + i * CC;
  float acc = 0.f;
  for (int m = 0; m < CC; ++m) acc += P[m] * Wsel[m * CC + j];
  Wc[which * CC * CC + i * CC + j] = acc;
  if (j == 0) {
    float b = ipb[which * CC + i];
    for (int m = 0; m < CC; ++m) b += P[m] * bsel[m];
    bc[which * CC + i] = b;
  }
}

// ---------------------------------------------------------------------------
// Kernel 4: q/k/v projection GEMM (32768x128)x(128x128) via f16 WMMA
// grid (512,1,3), block 256 (8 waves: 4 row strips x 2 col groups)
// ---------------------------------------------------------------------------
__global__ __launch_bounds__(256) void proj_gemm_kernel(
    const float* __restrict__ initfv, const float* __restrict__ comb,
    const float* __restrict__ Wc, const float* __restrict__ bc,
    float* __restrict__ qp, float* __restrict__ kp, float* __restrict__ vp) {
  const int z = blockIdx.z;
  const float* X  = (z == 0) ? initfv : comb;
  const float* Wz = Wc + z * CC * CC;
  const float* bz = bc + z * CC;
  float* Y = (z == 0) ? qp : (z == 1) ? kp : vp;

  const int tid  = threadIdx.x;
  const int wave = tid >> 5, lane = tid & 31;
  const int lm   = lane & 15, hi = lane >> 4;
  const int gr0  = blockIdx.x * 64 + (wave & 3) * 16;

  // A fragments (f16 16x32): lanes 0-15 hold K 0..7 / 16..23; lanes 16-31 hold K 8..15 / 24..31
  v16h afr[4];
  const float* xrow = X + (size_t)(gr0 + lm) * CC;
#pragma unroll
  for (int kk = 0; kk < 4; ++kk) {
    const int k0 = kk * 32;
#pragma unroll
    for (int pp = 0; pp < 8; ++pp) {
      afr[kk][pp]     = (_Float16)xrow[k0 + hi * 8 + pp];
      afr[kk][pp + 8] = (_Float16)xrow[k0 + 16 + hi * 8 + pp];
    }
  }
#pragma unroll
  for (int i = 0; i < 4; ++i) {
    const int col0 = ((wave >> 2) * 4 + i) * 16;
    const float* wrow = Wz + (size_t)(col0 + lm) * CC;   // B[kk][nn] = Wc[col][kk]
    v8f acc = {};
#pragma unroll
    for (int kk = 0; kk < 4; ++kk) {
      const int k0 = kk * 32 + hi * 8;
      v16h bfr;
#pragma unroll
      for (int pp = 0; pp < 16; ++pp) bfr[pp] = (_Float16)wrow[k0 + (pp & 7) + (pp >> 3) * 16];
      acc = __builtin_amdgcn_wmma_f32_16x16x32_f16(false, afr[kk], false, bfr,
                                                   (short)0, acc, false, false);
    }
    const int col = col0 + lm;
    const float bias = bz[col];
#pragma unroll
    for (int vv = 0; vv < 8; ++vv) {
      const int m = vv + hi * 8;
      Y[(size_t)(gr0 + m) * CC + col] = acc[vv] + bias;
    }
  }
}

// ---------------------------------------------------------------------------
// Kernel 5: cross-batch attention (4x4 per (k,head)) — scalar VALU
// ---------------------------------------------------------------------------
__global__ __launch_bounds__(256) void attn_kernel(const float* __restrict__ qp,
                                                   const float* __restrict__ kp,
                                                   const float* __restrict__ vp,
                                                   float* __restrict__ ao) {
  const int pair = blockIdx.x * 256 + threadIdx.x;  // K*NHD = 65536
  const int k = pair >> 3, h = pair & 7;
  const int base = k * CC + h * HD;
  float s[4][4];
#pragma unroll
  for (int nn = 0; nn < 4; ++nn) {
    const float* qr = qp + (size_t)nn * KK * CC + base;
#pragma unroll
    for (int mm = 0; mm < 4; ++mm) {
      const float* kr = kp + (size_t)mm * KK * CC + base;
      float acc = 0.f;
#pragma unroll
      for (int d = 0; d < HD; ++d) acc += qr[d] * kr[d];
      s[nn][mm] = acc * 0.25f;                         // 1/sqrt(16)
    }
  }
#pragma unroll
  for (int nn = 0; nn < 4; ++nn) {
    float mx = s[nn][0];
    for (int mm = 1; mm < 4; ++mm) mx = fmaxf(mx, s[nn][mm]);
    float den = 0.f;
    for (int mm = 0; mm < 4; ++mm) { s[nn][mm] = expf(s[nn][mm] - mx); den += s[nn][mm]; }
    const float inv = 1.f / den;
    for (int mm = 0; mm < 4; ++mm) s[nn][mm] *= inv;
  }
#pragma unroll
  for (int nn = 0; nn < 4; ++nn)
    for (int d = 0; d < HD; ++d) {
      float acc = 0.f;
#pragma unroll
      for (int mm = 0; mm < 4; ++mm) acc += s[nn][mm] * vp[(size_t)mm * KK * CC + base + d];
      ao[(size_t)nn * KK * CC + base + d] = acc;
    }
}

// ---------------------------------------------------------------------------
// Kernel 6: out projection GEMM + bias + residual (f16 WMMA)
// ---------------------------------------------------------------------------
__global__ __launch_bounds__(256) void out_gemm_kernel(const float* __restrict__ ao,
                                                       const float* __restrict__ outw,
                                                       const float* __restrict__ outb,
                                                       const float* __restrict__ initfv,
                                                       float* __restrict__ out) {
  const int tid  = threadIdx.x;
  const int wave = tid >> 5, lane = tid & 31;
  const int lm   = lane & 15, hi = lane >> 4;
  const int gr0  = blockIdx.x * 64 + (wave & 3) * 16;
  v16h afr[4];
  const float* xrow = ao + (size_t)(gr0 + lm) * CC;
#pragma unroll
  for (int kk = 0; kk < 4; ++kk) {
    const int k0 = kk * 32;
#pragma unroll
    for (int pp = 0; pp < 8; ++pp) {
      afr[kk][pp]     = (_Float16)xrow[k0 + hi * 8 + pp];
      afr[kk][pp + 8] = (_Float16)xrow[k0 + 16 + hi * 8 + pp];
    }
  }
#pragma unroll
  for (int i = 0; i < 4; ++i) {
    const int col0 = ((wave >> 2) * 4 + i) * 16;
    const float* wrow = outw + (size_t)(col0 + lm) * CC;
    v8f acc = {};
#pragma unroll
    for (int kk = 0; kk < 4; ++kk) {
      const int k0 = kk * 32 + hi * 8;
      v16h bfr;
#pragma unroll
      for (int pp = 0; pp < 16; ++pp) bfr[pp] = (_Float16)wrow[k0 + (pp & 7) + (pp >> 3) * 16];
      acc = __builtin_amdgcn_wmma_f32_16x16x32_f16(false, afr[kk], false, bfr,
                                                   (short)0, acc, false, false);
    }
    const int col = col0 + lm;
    const float bias = outb[col];
#pragma unroll
    for (int vv = 0; vv < 8; ++vv) {
      const int m = vv + hi * 8;
      const size_t idx = (size_t)(gr0 + m) * CC + col;
      out[idx] = acc[vv] + bias + initfv[idx];
    }
  }
}

// ---------------------------------------------------------------------------
extern "C" void kernel_launch(void* const* d_in, const int* in_sizes, int n_in,
                              void* d_out, int out_size, void* d_ws, size_t ws_size,
                              hipStream_t stream) {
  const float* fm   = (const float*)d_in[0];
  const float* xyz  = (const float*)d_in[1];
  const float* Wq   = (const float*)d_in[2];
  const float* bq   = (const float*)d_in[3];
  const float* Wk   = (const float*)d_in[4];
  const float* bk   = (const float*)d_in[5];
  const float* Wv   = (const float*)d_in[6];
  const float* bv   = (const float*)d_in[7];
  const float* ipw  = (const float*)d_in[8];
  const float* ipb  = (const float*)d_in[9];
  const float* outw = (const float*)d_in[10];
  const float* outb = (const float*)d_in[11];
  float* out = (float*)d_out;
  float* ws  = (float*)d_ws;

  float* simf   = ws;                       // N*C*D*H*W      = 8388608
  float* initfv = simf   + 8388608;         // N*K*C          = 4194304
  float* comb   = initfv + 4194304;
  float* qp     = comb   + 4194304;
  float* kp     = qp     + 4194304;
  float* vp     = kp     + 4194304;
  float* ao     = vp     + 4194304;
  float* Wc     = ao     + 4194304;         // 3*128*128
  float* bc     = Wc     + 3 * CC * CC;     // 3*128

  const size_t lds1 =
      (size_t)(32 * 1024 + 2 * 128 * BROW + 32 * 8 * 5 + 32 * 5) * sizeof(float) +
      (size_t)(32 * 8 * 5 + 32 * 5) * sizeof(int);

  sim_search_kernel<<<dim3(32, 64), 256, lds1, stream>>>(fm, simf);
  sample_kernel<<<dim3(KK, NB), 128, 0, stream>>>(fm, simf, xyz, initfv, comb);
  compose_kernel<<<dim3(384), 128, 0, stream>>>(Wq, bq, Wk, bk, Wv, bv, ipw, ipb, Wc, bc);
  proj_gemm_kernel<<<dim3(512, 1, 3), 256, 0, stream>>>(initfv, comb, Wc, bc, qp, kp, vp);
  attn_kernel<<<dim3(256), 256, 0, stream>>>(qp, kp, vp, ao);
  out_gemm_kernel<<<dim3(512), 256, 0, stream>>>(ao, outw, outb, initfv, out);
}
